// GATEncoderWithEdgeAttr_33328946217096
// MI455X (gfx1250) — matrix operand order repaired and checked by reading, not compiled, not run on hardware
//
#include <hip/hip_runtime.h>
#include <hip/hip_bf16.h>
#include <math.h>

// ---------------------------------------------------------------------------
// Problem constants (from reference)
// ---------------------------------------------------------------------------
#define N_NODES 4096
#define N_EDGES 131072
#define IN_F    128
#define ED_F    16
#define HID_F   64
#define HEADS   4
#define HC      256           // HEADS * HID
#define ETOT    (N_EDGES + N_NODES)

typedef __attribute__((ext_vector_type(16))) __bf16 v16bf;
typedef __attribute__((ext_vector_type(8)))  __bf16 v8bf;
typedef __attribute__((ext_vector_type(8)))  float  v8f;
typedef int vi4 __attribute__((vector_size(16)));   // int4 as expected by async-LDS builtin

#define DEV __device__ __forceinline__
#define AS1 __attribute__((address_space(1)))
#define AS3 __attribute__((address_space(3)))

#if __has_builtin(__builtin_amdgcn_global_load_async_to_lds_b128) && \
    __has_builtin(__builtin_amdgcn_s_wait_asynccnt)
#define USE_ASYNC_LDS 1
#else
#define USE_ASYNC_LDS 0
#endif

static DEV __bf16 to_bf16(float f) {
  unsigned u = __builtin_bit_cast(unsigned, f);
  u += 0x7FFFu + ((u >> 16) & 1u);          // round-to-nearest-even
  unsigned short h = (unsigned short)(u >> 16);
  return __builtin_bit_cast(__bf16, h);
}
// order-preserving float<->uint mapping for atomic max on floats
static DEV unsigned f2ord(float f) {
  unsigned u = __builtin_bit_cast(unsigned, f);
  return (u & 0x80000000u) ? ~u : (u | 0x80000000u);
}
static DEV float ord2f(unsigned u) {
  unsigned v = (u & 0x80000000u) ? (u ^ 0x80000000u) : ~u;
  return __builtin_bit_cast(float, v);
}
static DEV v16bf cat16(v8bf lo, v8bf hi) {
  return __builtin_shufflevector(lo, hi, 0, 1, 2, 3, 4, 5, 6, 7,
                                 8, 9, 10, 11, 12, 13, 14, 15);
}

// ---------------------------------------------------------------------------
// WMMA bf16 GEMM:  C[M,N] = A[M,K] @ B[K,N] (+bias[col]) (+relu)
// B is supplied TRANSPOSED: BT[N,K] so LDS fragments are contiguous b128 runs.
// block = 256 threads = 8 waves; macro tile 128x64; K-tile 64; each wave owns
// a 32x32 tile => 8 v_wmma per staged K-tile. Global->LDS staging via async
// LDS loads (ASYNCcnt) when the toolchain exposes them.
// Requires: M % 128 == 0, N % 64 == 0, K % 64 == 0.
// ---------------------------------------------------------------------------
__global__ void k_gemm_bf16(const __bf16* __restrict__ A, int lda,
                            const __bf16* __restrict__ BT, int ldbt,
                            const float* __restrict__ bias,
                            float* __restrict__ C, int ldc,
                            int M, int N, int K, int relu) {
  __shared__ __align__(16) __bf16 As[128][64];   // row-major [m][k]   16KB
  __shared__ __align__(16) __bf16 Bs[64][64];    // transposed [n][k]   8KB
  const int tid  = threadIdx.x;
  const int wave = tid >> 5;
  const int lane = tid & 31;
  const int lrow = lane & 15;          // row/col within 16x16 tile
  const int half = lane >> 4;          // lane half selects K-group / M-group
  const int wy = wave >> 1;            // 0..3 : 32-row slice
  const int wx = wave & 1;             // 0..1 : 32-col slice

  v8f acc00 = {}, acc01 = {}, acc10 = {}, acc11 = {};

  for (int k0 = 0; k0 < K; k0 += 64) {
#if USE_ASYNC_LDS
    // stage A tile 128x64 (4 x b128/thread) + B tile 64x64 (2 x b128/thread)
    // directly into LDS, no VGPR round-trip; tracked by ASYNCcnt.
#pragma unroll
    for (int q = 0; q < 4; ++q) {
      int idx = tid * 4 + q;
      int r = idx >> 3, c2 = (idx & 7) * 8;
      __builtin_amdgcn_global_load_async_to_lds_b128(
          (AS1 vi4*)(void*)(A + (size_t)(blockIdx.y * 128 + r) * lda + k0 + c2),
          (AS3 vi4*)(void*)&As[r][c2], 0, 0);
    }
#pragma unroll
    for (int q = 0; q < 2; ++q) {
      int idx = tid * 2 + q;
      int r = idx >> 3, c2 = (idx & 7) * 8;
      __builtin_amdgcn_global_load_async_to_lds_b128(
          (AS1 vi4*)(void*)(BT + (size_t)(blockIdx.x * 64 + r) * ldbt + k0 + c2),
          (AS3 vi4*)(void*)&Bs[r][c2], 0, 0);
    }
    __builtin_amdgcn_s_wait_asynccnt(0);
#else
    float4 ta[4]; float4 tb[2];
#pragma unroll
    for (int q = 0; q < 4; ++q) {
      int idx = tid * 4 + q;
      int r = idx >> 3, c2 = (idx & 7) * 8;
      ta[q] = *reinterpret_cast<const float4*>(
          A + (size_t)(blockIdx.y * 128 + r) * lda + k0 + c2);
    }
#pragma unroll
    for (int q = 0; q < 2; ++q) {
      int idx = tid * 2 + q;
      int r = idx >> 3, c2 = (idx & 7) * 8;
      tb[q] = *reinterpret_cast<const float4*>(
          BT + (size_t)(blockIdx.x * 64 + r) * ldbt + k0 + c2);
    }
#pragma unroll
    for (int q = 0; q < 4; ++q) {
      int idx = tid * 4 + q;
      int r = idx >> 3, c2 = (idx & 7) * 8;
      *reinterpret_cast<float4*>(&As[r][c2]) = ta[q];
    }
#pragma unroll
    for (int q = 0; q < 2; ++q) {
      int idx = tid * 2 + q;
      int r = idx >> 3, c2 = (idx & 7) * 8;
      *reinterpret_cast<float4*>(&Bs[r][c2]) = tb[q];
    }
#endif
    if (k0 + 64 < K) {
      __builtin_prefetch(A + (size_t)(blockIdx.y * 128 + tid / 2) * lda + k0 + 64, 0, 1);
      __builtin_prefetch(BT + (size_t)(blockIdx.x * 64 + tid / 4) * ldbt + k0 + 64, 0, 1);
    }
    __syncthreads();

    const int a0 = wy * 32 + lrow, a1 = a0 + 16;
    const int b0 = wx * 32 + lrow, b1 = b0 + 16;
#pragma unroll
    for (int kk = 0; kk < 2; ++kk) {
      const int kb = kk * 32;
      // A fragments: lane (m,half) elements are K in [8h,8h+7] U [16+8h,23+8h]
      v16bf af0 = cat16(*(const v8bf*)&As[a0][kb + 8 * half],
                        *(const v8bf*)&As[a0][kb + 16 + 8 * half]);
      v16bf af1 = cat16(*(const v8bf*)&As[a1][kb + 8 * half],
                        *(const v8bf*)&As[a1][kb + 16 + 8 * half]);
      // B fragments: lane (n,half) elements are K in [8h,8h+7] U [16+8h,23+8h]
      v16bf bf0 = cat16(*(const v8bf*)&Bs[b0][kb + 8 * half],
                        *(const v8bf*)&Bs[b0][kb + 16 + 8 * half]);
      v16bf bf1 = cat16(*(const v8bf*)&Bs[b1][kb + 8 * half],
                        *(const v8bf*)&Bs[b1][kb + 16 + 8 * half]);
      acc00 = __builtin_amdgcn_wmma_f32_16x16x32_bf16(false, af0, false, bf0,
                                                      (short)0, acc00, false, false);
      acc01 = __builtin_amdgcn_wmma_f32_16x16x32_bf16(false, af0, false, bf1,
                                                      (short)0, acc01, false, false);
      acc10 = __builtin_amdgcn_wmma_f32_16x16x32_bf16(false, af1, false, bf0,
                                                      (short)0, acc10, false, false);
      acc11 = __builtin_amdgcn_wmma_f32_16x16x32_bf16(false, af1, false, bf1,
                                                      (short)0, acc11, false, false);
    }
    __syncthreads();
  }

  const int gn0 = blockIdx.x * 64 + wx * 32 + lrow;
  const int gn1 = gn0 + 16;
  const float bb0 = bias ? bias[gn0] : 0.0f;
  const float bb1 = bias ? bias[gn1] : 0.0f;
#pragma unroll
  for (int i = 0; i < 8; ++i) {
    // C/D layout: VGPR i: lanes0-15 M=i, lanes16-31 M=i+8
    int gm0 = blockIdx.y * 128 + wy * 32 + i + 8 * half;
    int gm1 = gm0 + 16;
    float v00 = acc00[i] + bb0, v01 = acc01[i] + bb1;
    float v10 = acc10[i] + bb0, v11 = acc11[i] + bb1;
    if (relu) {
      v00 = fmaxf(v00, 0.0f); v01 = fmaxf(v01, 0.0f);
      v10 = fmaxf(v10, 0.0f); v11 = fmaxf(v11, 0.0f);
    }
    C[(size_t)gm0 * ldc + gn0] = v00;
    C[(size_t)gm0 * ldc + gn1] = v01;
    C[(size_t)gm1 * ldc + gn0] = v10;
    C[(size_t)gm1 * ldc + gn1] = v11;
  }
}

// ---------------------------------------------------------------------------
// Utility kernels
// ---------------------------------------------------------------------------
__global__ void k_zero_f32(float* p, long long n) {
  long long t = (long long)blockIdx.x * 256 + threadIdx.x;
  if (t < n) p[t] = 0.0f;
}
__global__ void k_copy_f32(const float* __restrict__ s, float* __restrict__ d,
                           long long n) {
  long long t = (long long)blockIdx.x * 256 + threadIdx.x;
  if (t < n) d[t] = s[t];
}
__global__ void k_f32_to_bf16(const float* __restrict__ s,
                              __bf16* __restrict__ d, long long n) {
  long long t = (long long)blockIdx.x * 256 + threadIdx.x;
  if (t < n) d[t] = to_bf16(s[t]);
}
// s[rows x cols] (f32) -> d[cols x rows] (bf16)   (for GEMM B operands)
__global__ void k_f32_to_bf16_T(const float* __restrict__ s,
                                __bf16* __restrict__ d, int rows, int cols) {
  long long t = (long long)blockIdx.x * 256 + threadIdx.x;
  if (t >= (long long)rows * cols) return;
  int r = (int)(t / cols), c = (int)(t % cols);
  d[(size_t)c * rows + r] = to_bf16(s[t]);
}
__global__ void k_colmean(const float* __restrict__ src, int rows, int cols,
                          float* __restrict__ mean) {
  __shared__ float sm[256];
  int col = blockIdx.x, t = threadIdx.x;
  float a = 0.0f;
  for (int r = t; r < rows; r += 256) a += src[(size_t)r * cols + col];
  sm[t] = a; __syncthreads();
  for (int s = 128; s > 0; s >>= 1) {
    if (t < s) sm[t] += sm[t + s];
    __syncthreads();
  }
  if (t == 0) mean[col] = sm[0] / (float)rows;
}

// ---------------------------------------------------------------------------
// GATv2 kernels
// ---------------------------------------------------------------------------
__global__ void k_gat_score(const float* __restrict__ xl,
                            const float* __restrict__ xr,
                            const int* __restrict__ ei,
                            const float* __restrict__ edge_attr,
                            const float* __restrict__ ea_mean,
                            const float* __restrict__ We,
                            const float* __restrict__ att,
                            float* __restrict__ score,
                            unsigned* __restrict__ smax) {
  long long t = (long long)blockIdx.x * 256 + threadIdx.x;
  if (t >= (long long)ETOT * HEADS) return;
  int i = (int)(t / HEADS), hh = (int)(t % HEADS);
  int s, d;
  float ea[ED_F];
  if (i < N_EDGES) {
    s = ei[i]; d = ei[N_EDGES + i];
#pragma unroll
    for (int q = 0; q < ED_F; ++q) ea[q] = edge_attr[(size_t)i * ED_F + q];
  } else {
    s = i - N_EDGES; d = i - N_EDGES;
#pragma unroll
    for (int q = 0; q < ED_F; ++q) ea[q] = ea_mean[q];
  }
  float acc = 0.0f;
  for (int cch = 0; cch < HID_F; ++cch) {
    int col = hh * HID_F + cch;
    float ec = 0.0f;
#pragma unroll
    for (int q = 0; q < ED_F; ++q) ec += ea[q] * We[(size_t)q * HC + col];
    float m = xl[(size_t)s * HC + col] + xr[(size_t)d * HC + col] + ec;
    m = (m > 0.0f) ? m : 0.2f * m;   // leaky relu 0.2
    acc += m * att[col];
  }
  score[t] = acc;
  atomicMax(&smax[d * HEADS + hh], f2ord(acc));
}
__global__ void k_gat_exp(const float* __restrict__ score,
                          const int* __restrict__ ei,
                          const unsigned* __restrict__ smax,
                          float* __restrict__ ex, float* __restrict__ den) {
  long long t = (long long)blockIdx.x * 256 + threadIdx.x;
  if (t >= (long long)ETOT * HEADS) return;
  int i = (int)(t / HEADS), hh = (int)(t % HEADS);
  int d = (i < N_EDGES) ? ei[N_EDGES + i] : (i - N_EDGES);
  float e = expf(score[t] - ord2f(smax[d * HEADS + hh]));
  ex[t] = e;
  atomicAdd(&den[d * HEADS + hh], e);
}
__global__ void k_gat_aggr(const float* __restrict__ ex,
                           const float* __restrict__ den,
                           const int* __restrict__ ei,
                           const float* __restrict__ xl,
                           float* __restrict__ out) {
  long long t = (long long)blockIdx.x * 256 + threadIdx.x;
  if (t >= (long long)ETOT * HEADS) return;
  int i = (int)(t / HEADS), hh = (int)(t % HEADS);
  int s, d;
  if (i < N_EDGES) { s = ei[i]; d = ei[N_EDGES + i]; }
  else { s = i - N_EDGES; d = s; }
  float alpha = ex[t] / den[d * HEADS + hh];
  for (int cch = 0; cch < HID_F; ++cch) {
    int col = hh * HID_F + cch;
    atomicAdd(&out[(size_t)d * HC + col], xl[(size_t)s * HC + col] * alpha);
  }
}
__global__ void k_gat_finish(const float* __restrict__ go,
                             const float* __restrict__ bias,
                             float* __restrict__ h) {
  long long t = (long long)blockIdx.x * 256 + threadIdx.x;
  if (t >= (long long)N_NODES * HC) return;
  float v = go[t] + bias[t % HC];
  h[t] = (v > 0.0f) ? v : expm1f(v);   // ELU
}
__global__ void k_bn_stats(const float* __restrict__ h, int rows,
                           float* __restrict__ mu, float* __restrict__ var) {
  __shared__ float s1[256], s2[256];
  int col = blockIdx.x, t = threadIdx.x;
  float a = 0.0f, b = 0.0f;
  for (int r = t; r < rows; r += 256) {
    float x = h[(size_t)r * HC + col];
    a += x; b += x * x;
  }
  s1[t] = a; s2[t] = b; __syncthreads();
  for (int s = 128; s > 0; s >>= 1) {
    if (t < s) { s1[t] += s1[t + s]; s2[t] += s2[t + s]; }
    __syncthreads();
  }
  if (t == 0) {
    float m = s1[0] / (float)rows;
    mu[col] = m;
    var[col] = s2[0] / (float)rows - m * m;
  }
}
__global__ void k_bn_apply(float* __restrict__ h, const float* __restrict__ mu,
                           const float* __restrict__ var,
                           const float* __restrict__ g,
                           const float* __restrict__ b) {
  long long t = (long long)blockIdx.x * 256 + threadIdx.x;
  if (t >= (long long)N_NODES * HC) return;
  int c = (int)(t % HC);
  h[t] = (h[t] - mu[c]) * rsqrtf(var[c] + 1e-5f) * g[c] + b[c];
}

// ---------------------------------------------------------------------------
// Dense-graph kernels
// ---------------------------------------------------------------------------
__global__ void k_build_A(const int* __restrict__ ei, float* __restrict__ A,
                          int n) {
  int e = blockIdx.x * 256 + threadIdx.x;
  if (e >= N_EDGES) return;
  atomicAdd(&A[(size_t)ei[N_EDGES + e] * n + ei[e]], 1.0f);
}
__global__ void k_rowsum_rsqrt(const float* __restrict__ A, int n,
                               float* __restrict__ dis) {
  __shared__ float sm[256];
  int row = blockIdx.x, t = threadIdx.x;
  float a = 0.0f;
  for (int j = t; j < n; j += 256) a += A[(size_t)row * n + j];
  sm[t] = a; __syncthreads();
  for (int s = 128; s > 0; s >>= 1) {
    if (t < s) sm[t] += sm[t + s];
    __syncthreads();
  }
  if (t == 0) {
    float d = sm[0] + 2.0f;                  // improved self-loop: +2I
    dis[row] = (d > 0.0f) ? rsqrtf(d) : 0.0f;
  }
}
__global__ void k_norm_adj_bf16(const float* __restrict__ A,
                                const float* __restrict__ dis,
                                __bf16* __restrict__ out, int n) {
  long long t = (long long)blockIdx.x * 256 + threadIdx.x;
  if (t >= (long long)n * n) return;
  int i = (int)(t / n), j = (int)(t % n);
  float a = A[t] + ((i == j) ? 2.0f : 0.0f);
  out[t] = to_bf16(dis[i] * a * dis[j]);
}
// A1 = A*(1-I)+I ; writes both row-major (for A operand) and transposed (B op)
__global__ void k_augment_bf16(const float* __restrict__ A,
                               __bf16* __restrict__ out,
                               __bf16* __restrict__ outT, int n) {
  long long t = (long long)blockIdx.x * 256 + threadIdx.x;
  if (t >= (long long)n * n) return;
  int i = (int)(t / n), j = (int)(t % n);
  __bf16 b = to_bf16((i == j) ? 1.0f : A[t]);
  out[t] = b;
  outT[(size_t)j * n + i] = b;
}
__global__ void k_zero_diag(float* __restrict__ A, int n) {
  int i = blockIdx.x * 256 + threadIdx.x;
  if (i < n) A[(size_t)i * n + i] = 0.0f;
}

// ---------------------------------------------------------------------------
// TopK pooling kernels
// ---------------------------------------------------------------------------
__global__ void k_vec_norm(const float* __restrict__ p, float* __restrict__ out) {
  __shared__ float s[64];
  int t = threadIdx.x;
  float v = p[t];
  s[t] = v * v; __syncthreads();
  for (int k = 32; k > 0; k >>= 1) {
    if (t < k) s[t] += s[t + k];
    __syncthreads();
  }
  if (t == 0) out[0] = sqrtf(s[0]);
}
__global__ void k_pool_score(const float* __restrict__ x,
                             const float* __restrict__ p,
                             const float* __restrict__ pn,
                             float* __restrict__ s, int n) {
  int i = blockIdx.x * 256 + threadIdx.x;
  if (i >= n) return;
  float d = 0.0f;
#pragma unroll
  for (int c = 0; c < HID_F; ++c) d += x[(size_t)i * HID_F + c] * p[c];
  s[i] = tanhf(d / pn[0]);
}
__global__ void k_rank_select(const float* __restrict__ s, int n, int k,
                              int* __restrict__ perm, float* __restrict__ vals) {
  int i = blockIdx.x * 256 + threadIdx.x;
  if (i >= n) return;
  float si = s[i];
  int r = 0;
  for (int j = 0; j < n; ++j) {
    float sj = s[j];
    r += (sj > si) || (sj == si && j < i);   // stable descending rank
  }
  if (r < k) { perm[r] = i; vals[r] = si; }
}
__global__ void k_gather_A(const float* __restrict__ Ain, int n,
                           const int* __restrict__ perm,
                           float* __restrict__ Aout, int k) {
  long long t = (long long)blockIdx.x * 256 + threadIdx.x;
  if (t >= (long long)k * k) return;
  int a = (int)(t / k), b = (int)(t % k);
  Aout[t] = Ain[(size_t)perm[a] * n + perm[b]];
}
__global__ void k_gather_x(const float* __restrict__ x,
                           const int* __restrict__ perm,
                           const float* __restrict__ vals,
                           float* __restrict__ xout, int k) {
  long long t = (long long)blockIdx.x * 256 + threadIdx.x;
  if (t >= (long long)k * HID_F) return;
  int a = (int)(t / HID_F), c = (int)(t % HID_F);
  xout[t] = x[(size_t)perm[a] * HID_F + c] * vals[a];
}
__global__ void k_scatter_add(const float* __restrict__ cur,
                              const int* __restrict__ perm,
                              float* __restrict__ dst, int k) {
  long long t = (long long)blockIdx.x * 256 + threadIdx.x;
  if (t >= (long long)k * HID_F) return;
  int a = (int)(t / HID_F), c = (int)(t % HID_F);
  dst[(size_t)perm[a] * HID_F + c] += cur[t];
}

// ---------------------------------------------------------------------------
// Host orchestration
// ---------------------------------------------------------------------------
extern "C" void kernel_launch(void* const* d_in, const int* in_sizes, int n_in,
                              void* d_out, int out_size, void* d_ws, size_t ws_size,
                              hipStream_t stream) {
  (void)in_sizes; (void)n_in; (void)out_size; (void)ws_size;
  const float* x         = (const float*)d_in[0];
  const int*   ei        = (const int*)  d_in[1];
  const float* edge_attr = (const float*)d_in[2];
  const float* Wl = (const float*)d_in[3];  const float* bl = (const float*)d_in[4];
  const float* Wr = (const float*)d_in[5];  const float* br = (const float*)d_in[6];
  const float* We = (const float*)d_in[7];
  const float* att = (const float*)d_in[8];
  const float* bias_gat = (const float*)d_in[9];
  const float* gamma = (const float*)d_in[10];
  const float* beta  = (const float*)d_in[11];
  const float* Wd[4] = {(const float*)d_in[12], (const float*)d_in[14],
                        (const float*)d_in[16], (const float*)d_in[18]};
  const float* bd[4] = {(const float*)d_in[13], (const float*)d_in[15],
                        (const float*)d_in[17], (const float*)d_in[19]};
  const float* pv[3] = {(const float*)d_in[20], (const float*)d_in[21],
                        (const float*)d_in[22]};
  const float* Wu[3] = {(const float*)d_in[23], (const float*)d_in[25],
                        (const float*)d_in[27]};
  const float* bu[3] = {(const float*)d_in[24], (const float*)d_in[26],
                        (const float*)d_in[28]};
  float* outp = (float*)d_out;

  // bump allocator over workspace
  size_t off = 0;
  auto alloc = [&](size_t bytes) -> char* {
    off = (off + 255) & ~(size_t)255;
    char* p = (char*)d_ws + off;
    off += bytes;
    return p;
  };
  const size_t NN = (size_t)N_NODES * N_NODES;
  __bf16* xh    = (__bf16*)alloc((size_t)N_NODES * IN_F * 2);
  __bf16* WlT   = (__bf16*)alloc((size_t)IN_F * HC * 2);   // [HC x IN]
  __bf16* WrT   = (__bf16*)alloc((size_t)IN_F * HC * 2);
  float*  xl    = (float*) alloc((size_t)N_NODES * HC * 4);
  float*  xr    = (float*) alloc((size_t)N_NODES * HC * 4);
  float*  eam   = (float*) alloc(ED_F * 4);
  float*  score = (float*) alloc((size_t)ETOT * HEADS * 4);
  float*  exb   = (float*) alloc((size_t)ETOT * HEADS * 4);
  unsigned* smax= (unsigned*)alloc((size_t)N_NODES * HEADS * 4);
  float*  den   = (float*) alloc((size_t)N_NODES * HEADS * 4);
  float*  gout  = (float*) alloc((size_t)N_NODES * HC * 4);
  float*  hbuf  = (float*) alloc((size_t)N_NODES * HC * 4);
  float*  mu    = (float*) alloc(HC * 4);
  float*  var   = (float*) alloc(HC * 4);
  float*  A0    = (float*) alloc(NN * 4);              // As[0]
  float*  Aaug  = (float*) alloc(NN * 4);              // augment scratch
  float*  As1   = (float*) alloc((size_t)2048 * 2048 * 4);
  float*  As2   = (float*) alloc((size_t)1024 * 1024 * 4);
  float*  A3    = (float*) alloc((size_t)512 * 512 * 4);
  __bf16* Abf   = (__bf16*)alloc(NN * 2);              // bf16 A operand
  __bf16* AbfT  = (__bf16*)alloc(NN * 2);              // bf16 B operand (transposed)
  __bf16* Xbf   = (__bf16*)alloc((size_t)N_NODES * HC * 2);
  __bf16* WgT   = (__bf16*)alloc((size_t)HC * HID_F * 2);  // [64 x inF]
  float*  XW    = (float*) alloc((size_t)N_NODES * HID_F * 4);
  __bf16* XWbT  = (__bf16*)alloc((size_t)N_NODES * HID_F * 2); // [64 x n]
  float*  dis   = (float*) alloc(N_NODES * 4);
  float*  xs0   = (float*) alloc((size_t)N_NODES * HID_F * 4);
  float*  p0x   = (float*) alloc((size_t)2048 * HID_F * 4);
  float*  xs1   = (float*) alloc((size_t)2048 * HID_F * 4);
  float*  p1x   = (float*) alloc((size_t)1024 * HID_F * 4);
  float*  xs2   = (float*) alloc((size_t)1024 * HID_F * 4);
  float*  p2x   = (float*) alloc((size_t)512 * HID_F * 4);
  float*  curD  = (float*) alloc((size_t)512 * HID_F * 4);
  int*    perm1 = (int*)   alloc(2048 * 4);
  int*    perm2 = (int*)   alloc(1024 * 4);
  int*    perm3 = (int*)   alloc(512 * 4);
  float*  vals1 = (float*) alloc(2048 * 4);
  float*  vals2 = (float*) alloc(1024 * 4);
  float*  vals3 = (float*) alloc(512 * 4);
  float*  sbuf  = (float*) alloc(N_NODES * 4);
  float*  pn    = (float*) alloc(256);
  float*  rbuf  = (float*) alloc((size_t)N_NODES * HID_F * 4);
  float*  ubuf  = (float*) alloc((size_t)N_NODES * HID_F * 4);

  auto g1 = [](long long n) { return dim3((unsigned)((n + 255) / 256)); };
  auto zero = [&](float* p, long long n) {
    k_zero_f32<<<g1(n), 256, 0, stream>>>(p, n);
  };
  auto cvt = [&](const float* s, __bf16* d, long long n) {
    k_f32_to_bf16<<<g1(n), 256, 0, stream>>>(s, d, n);
  };
  auto cvtT = [&](const float* s, __bf16* d, int rows, int cols) {
    k_f32_to_bf16_T<<<g1((long long)rows * cols), 256, 0, stream>>>(s, d, rows,
                                                                    cols);
  };
  auto gemm = [&](const __bf16* A, int lda, const __bf16* BT, int ldbt,
                  const float* bias, float* C, int ldc, int M, int N, int K,
                  int relu) {
    dim3 grid(N / 64, M / 128);
    k_gemm_bf16<<<grid, 256, 0, stream>>>(A, lda, BT, ldbt, bias, C, ldc, M, N,
                                          K, relu);
  };
  // GCNConv(improved): out = D^-1/2 (A+2I) D^-1/2 (X W) + b  [optional relu]
  auto gcn = [&](const float* Ad, int n, const float* X, int inF,
                 const float* W, const float* b, float* out, int relu) {
    k_rowsum_rsqrt<<<dim3(n), 256, 0, stream>>>(Ad, n, dis);
    k_norm_adj_bf16<<<g1((long long)n * n), 256, 0, stream>>>(Ad, dis, Abf, n);
    cvt(X, Xbf, (long long)n * inF);
    cvtT(W, WgT, inF, HID_F);                       // [64 x inF]
    gemm(Xbf, inF, WgT, inF, nullptr, XW, HID_F, n, HID_F, inF, 0);
    cvtT(XW, XWbT, n, HID_F);                       // [64 x n]
    gemm(Abf, n, XWbT, n, b, out, HID_F, n, HID_F, n, relu);
  };
  auto augment = [&](const float* Ain, int n, float* Aout) {
    k_augment_bf16<<<g1((long long)n * n), 256, 0, stream>>>(Ain, Abf, AbfT, n);
    gemm(Abf, n, AbfT, n, nullptr, Aout, n, n, n, n, 0);
    k_zero_diag<<<g1(n), 256, 0, stream>>>(Aout, n);
  };
  auto pool = [&](const float* Aap, const float* xin, const float* p, int n,
                  float* Aout, float* xout, int* perm, float* vals) {
    int k = n / 2;
    k_vec_norm<<<1, 64, 0, stream>>>(p, pn);
    k_pool_score<<<g1(n), 256, 0, stream>>>(xin, p, pn, sbuf, n);
    k_rank_select<<<g1(n), 256, 0, stream>>>(sbuf, n, k, perm, vals);
    k_gather_A<<<g1((long long)k * k), 256, 0, stream>>>(Aap, n, perm, Aout, k);
    k_gather_x<<<g1((long long)k * HID_F), 256, 0, stream>>>(xin, perm, vals,
                                                             xout, k);
  };

  // ---- GATv2 ----
  cvt(x, xh, (long long)N_NODES * IN_F);
  cvtT(Wl, WlT, IN_F, HC);
  cvtT(Wr, WrT, IN_F, HC);
  gemm(xh, IN_F, WlT, IN_F, bl, xl, HC, N_NODES, HC, IN_F, 0);
  gemm(xh, IN_F, WrT, IN_F, br, xr, HC, N_NODES, HC, IN_F, 0);
  k_colmean<<<dim3(ED_F), 256, 0, stream>>>(edge_attr, N_EDGES, ED_F, eam);
  zero((float*)smax, (long long)N_NODES * HEADS);   // 0 == lowest ordered code
  zero(den, (long long)N_NODES * HEADS);
  zero(gout, (long long)N_NODES * HC);
  k_gat_score<<<g1((long long)ETOT * HEADS), 256, 0, stream>>>(
      xl, xr, ei, edge_attr, eam, We, att, score, smax);
  k_gat_exp<<<g1((long long)ETOT * HEADS), 256, 0, stream>>>(score, ei, smax,
                                                             exb, den);
  k_gat_aggr<<<g1((long long)ETOT * HEADS), 256, 0, stream>>>(exb, den, ei, xl,
                                                              gout);
  k_gat_finish<<<g1((long long)N_NODES * HC), 256, 0, stream>>>(gout, bias_gat,
                                                                hbuf);
  k_bn_stats<<<dim3(HC), 256, 0, stream>>>(hbuf, N_NODES, mu, var);
  k_bn_apply<<<g1((long long)N_NODES * HC), 256, 0, stream>>>(hbuf, mu, var,
                                                              gamma, beta);

  // ---- dense adjacency ----
  zero(A0, (long long)NN);
  k_build_A<<<g1(N_EDGES), 256, 0, stream>>>(ei, A0, N_NODES);

  // ---- GraphUNet down path ----
  gcn(A0, N_NODES, hbuf, HC, Wd[0], bd[0], xs0, 1);
  augment(A0, N_NODES, Aaug);
  pool(Aaug, xs0, pv[0], N_NODES, As1, p0x, perm1, vals1);
  gcn(As1, 2048, p0x, HID_F, Wd[1], bd[1], xs1, 1);
  augment(As1, 2048, Aaug);
  pool(Aaug, xs1, pv[1], 2048, As2, p1x, perm2, vals2);
  gcn(As2, 1024, p1x, HID_F, Wd[2], bd[2], xs2, 1);
  augment(As2, 1024, Aaug);
  pool(Aaug, xs2, pv[2], 1024, A3, p2x, perm3, vals3);
  gcn(A3, 512, p2x, HID_F, Wd[3], bd[3], curD, 1);

  // ---- GraphUNet up path (sum_res) ----
  // i=0: j=2: res=xs2 (1024), A=As2, perm=perm3
  k_copy_f32<<<g1((long long)1024 * HID_F), 256, 0, stream>>>(
      xs2, rbuf, (long long)1024 * HID_F);
  k_scatter_add<<<g1((long long)512 * HID_F), 256, 0, stream>>>(curD, perm3,
                                                                rbuf, 512);
  gcn(As2, 1024, rbuf, HID_F, Wu[0], bu[0], ubuf, 1);
  // i=1: j=1: res=xs1 (2048), A=As1, perm=perm2
  k_copy_f32<<<g1((long long)2048 * HID_F), 256, 0, stream>>>(
      xs1, rbuf, (long long)2048 * HID_F);
  k_scatter_add<<<g1((long long)1024 * HID_F), 256, 0, stream>>>(ubuf, perm2,
                                                                 rbuf, 1024);
  gcn(As1, 2048, rbuf, HID_F, Wu[1], bu[1], ubuf, 1);
  // i=2: j=0: res=xs0 (4096), A=A0, perm=perm1 ; final, no relu, to d_out
  k_copy_f32<<<g1((long long)N_NODES * HID_F), 256, 0, stream>>>(
      xs0, rbuf, (long long)N_NODES * HID_F);
  k_scatter_add<<<g1((long long)2048 * HID_F), 256, 0, stream>>>(ubuf, perm1,
                                                                 rbuf, 2048);
  gcn(A0, N_NODES, rbuf, HID_F, Wu[2], bu[2], outp, 0);
}
